// CustomMultiHeadAttention_54554674593898
// MI455X (gfx1250) — matrix operand order repaired and checked by reading, not compiled, not run on hardware
//
#include <hip/hip_runtime.h>
#include <hip/hip_bf16.h>

typedef __attribute__((ext_vector_type(16))) _Float16 v16h;
typedef __attribute__((ext_vector_type(8)))  _Float16 v8h;
typedef __attribute__((ext_vector_type(8)))  float    v8f;
typedef __attribute__((ext_vector_type(4)))  int      v4i;

union HFrag { v16h v; v8h h[2]; };

#define WMMA_F16(A, Bf, C) \
    __builtin_amdgcn_wmma_f32_16x16x32_f16(false, (A), false, (Bf), (short)0, (C), false, false)

// ---- CDNA5 async global->LDS path (ASYNCcnt), with sync fallback ----------
#if defined(__has_builtin)
#  if __has_builtin(__builtin_amdgcn_global_load_async_to_lds_b128) && \
      __has_builtin(__builtin_amdgcn_s_wait_asynccnt)
#    define CDNA5_ASYNC 1
#  endif
#endif
#ifndef CDNA5_ASYNC
#  define CDNA5_ASYNC 0
#endif

__device__ __forceinline__ void async_copy_b128(const void* g, void* l) {
#if CDNA5_ASYNC
    __builtin_amdgcn_global_load_async_to_lds_b128(
        (__attribute__((address_space(1))) v4i*)(g),
        (__attribute__((address_space(3))) v4i*)(l),
        0, 0);
#else
    *(v8h*)l = *(const v8h*)g;
#endif
}
__device__ __forceinline__ void wait_async() {
#if CDNA5_ASYNC
    __builtin_amdgcn_s_wait_asynccnt(0);
#endif
}

static constexpr int E   = 512;
static constexpr int H   = 8;
static constexpr int Dh  = 64;
static constexpr int S   = 4096;
static constexpr int BSZ = 2;
static constexpr int M   = BSZ * S; // 8192

// ---------------------------------------------------------------------------
// Kernel 1: QKV projections.  C = x @ W + b, written as f16.
//   mat 0 -> Q  [b,h,s,d] (scaled by 1/sqrt(Dh))
//   mat 1 -> K  [b,h,s,d]
//   mat 2 -> V^T [b,h,d,s]   (so P@V B-fragments are contiguous)
// Block tile 128x64, 8 waves, per-wave 32x32 (2x2 WMMA subtiles), K-step 32.
// (f32->f16 conversion happens in flight, so staging stays synchronous.)
// ---------------------------------------------------------------------------
__global__ __launch_bounds__(256)
void qkv_kernel(const float* __restrict__ x,
                const float* __restrict__ Wq, const float* __restrict__ bq,
                const float* __restrict__ Wk, const float* __restrict__ bk,
                const float* __restrict__ Wv, const float* __restrict__ bv,
                _Float16* __restrict__ qOut, _Float16* __restrict__ kOut,
                _Float16* __restrict__ vtOut)
{
    const int mat = blockIdx.z;
    const float* W    = (mat == 0) ? Wq : (mat == 1) ? Wk : Wv;
    const float* bias = (mat == 0) ? bq : (mat == 1) ? bk : bv;
    const int m0 = blockIdx.x * 128;
    const int n0 = blockIdx.y * 64;

    __shared__ __align__(16) _Float16 Xs[128][40]; // [row][k], 16B-multiple stride
    __shared__ __align__(16) _Float16 Ws[64][40];  // transposed: [n][k]

    const int tid  = threadIdx.x;
    const int lane = tid & 31;
    const int wave = tid >> 5;
    const int wr   = wave >> 1;   // 0..3 : row group
    const int wc   = wave & 1;    // 0..1 : col group
    const int l16  = lane & 15;
    const int hi   = lane >> 4;   // 0/1 : wave half

    v8f acc[2][2] = {};

    for (int kb = 0; kb < E; kb += 32) {
        __syncthreads();
        // --- stage X tile 128x32 fp32 -> f16 LDS ---
        {
            const int row = tid >> 1;
            const int kc  = (tid & 1) * 16;
            const float* src = x + (size_t)(m0 + row) * E + kb + kc;
#pragma unroll
            for (int i = 0; i < 4; ++i) {
                float4 f = *(const float4*)(src + i * 4);
                Xs[row][kc + i*4 + 0] = (_Float16)f.x;
                Xs[row][kc + i*4 + 1] = (_Float16)f.y;
                Xs[row][kc + i*4 + 2] = (_Float16)f.z;
                Xs[row][kc + i*4 + 3] = (_Float16)f.w;
            }
        }
        // --- stage W tile 32x64 fp32 -> f16 LDS, transposed to [n][k] ---
#pragma unroll
        for (int it = 0; it < 2; ++it) {
            const int u    = tid + it * 256;       // 0..511 float4 chunks
            const int krow = u >> 4;               // 0..31
            const int nc   = (u & 15) * 4;         // 0..60
            float4 f = *(const float4*)(W + (size_t)(kb + krow) * E + n0 + nc);
            Ws[nc + 0][krow] = (_Float16)f.x;
            Ws[nc + 1][krow] = (_Float16)f.y;
            Ws[nc + 2][krow] = (_Float16)f.z;
            Ws[nc + 3][krow] = (_Float16)f.w;
        }
        __syncthreads();

        // --- fragments per ISA layout ---
        HFrag a[2], b[2];
#pragma unroll
        for (int i = 0; i < 2; ++i) {
            const int row = wr * 32 + i * 16 + l16;
            a[i].h[0] = *(const v8h*)&Xs[row][hi * 8];
            a[i].h[1] = *(const v8h*)&Xs[row][16 + hi * 8];
        }
#pragma unroll
        for (int j = 0; j < 2; ++j) {
            const int col = wc * 32 + j * 16 + l16;
            b[j].h[0] = *(const v8h*)&Ws[col][hi * 8];
            b[j].h[1] = *(const v8h*)&Ws[col][16 + hi * 8];
        }
#pragma unroll
        for (int i = 0; i < 2; ++i)
#pragma unroll
            for (int j = 0; j < 2; ++j)
                acc[i][j] = WMMA_F16(a[i].v, b[j].v, acc[i][j]);
    }

    // --- epilogue: bias, (scale for Q), scatter to attention layouts ---
#pragma unroll
    for (int i = 0; i < 2; ++i)
#pragma unroll
        for (int j = 0; j < 2; ++j) {
            const int col = n0 + wc * 32 + j * 16 + l16;
            const float bv_ = bias[col];
            const int hh = col >> 6, d = col & 63;
#pragma unroll
            for (int r = 0; r < 8; ++r) {
                const int m = m0 + wr * 32 + i * 16 + r + hi * 8;
                float v = acc[i][j][r] + bv_;
                if (mat == 0) v *= 0.125f;  // 1/sqrt(Dh) folded into Q
                const int bI = m >> 12, s = m & (S - 1);
                if (mat == 2) {
                    vtOut[(((size_t)(bI * H + hh)) * Dh + d) * S + s] = (_Float16)v;
                } else {
                    _Float16* dst = (mat == 0) ? qOut : kOut;
                    dst[(((size_t)(bI * H + hh)) * S + s) * Dh + d] = (_Float16)v;
                }
            }
        }
}

// ---------------------------------------------------------------------------
// Kernel 2: flash attention. One block per (b*h, 128-query tile); each of the
// 8 waves owns 16 queries. K and V^T tiles are double-buffered in LDS and
// filled by async global->LDS DMA that overlaps the WMMA work of the
// previous tile.
// ---------------------------------------------------------------------------
__global__ __launch_bounds__(256)
void flash_kernel(const _Float16* __restrict__ q,
                  const _Float16* __restrict__ k,
                  const _Float16* __restrict__ vt,
                  _Float16* __restrict__ attn)
{
    const int bh = blockIdx.x;          // b*H + h
    const int qt = blockIdx.y;          // query tile (128 queries)
    const int bI = bh >> 3, hh = bh & 7;

    __shared__ __align__(16) _Float16 Ks[2][64][72];    // K tile [key][d]
    __shared__ __align__(16) _Float16 Vs[2][64][72];    // V^T tile [d][key]
    __shared__ __align__(16) _Float16 Ps[8][16][72];    // per-wave P patch

    const int tid  = threadIdx.x;
    const int lane = tid & 31;
    const int wave = tid >> 5;
    const int l16  = lane & 15;
    const int hi   = lane >> 4;
    const int qbase = qt * 128 + wave * 16;

    const _Float16* kBase = k  + (size_t)bh * S * Dh;   // [key][d]
    const _Float16* vBase = vt + (size_t)bh * Dh * S;   // [d][key]

    // issue async DMA for key-block kb into buffer buf (K: 512 chunks, V: 512)
    auto stage = [&](int kb, int buf) {
#pragma unroll
        for (int it = 0; it < 2; ++it) {
            const int u   = tid + it * 256;   // 0..511 16-byte chunks
            const int row = u >> 3;           // K: key row | V: d row
            const int c   = (u & 7) * 8;
            async_copy_b128(kBase + ((size_t)(kb * 64 + row)) * Dh + c,
                            &Ks[buf][row][c]);
            async_copy_b128(vBase + (size_t)row * S + kb * 64 + c,
                            &Vs[buf][row][c]);
        }
    };

    // Q fragments for this wave's 16 rows (resident for the whole loop)
    HFrag qa[2];
    {
        const _Float16* qrow = q + ((size_t)bh * S + qbase + l16) * Dh;
#pragma unroll
        for (int c = 0; c < 2; ++c) {
            qa[c].h[0] = *(const v8h*)(qrow + c * 32 + hi * 8);
            qa[c].h[1] = *(const v8h*)(qrow + c * 32 + 16 + hi * 8);
        }
    }

    float m_run[8], l_run[8];
    v8f oacc[4] = {};
#pragma unroll
    for (int r = 0; r < 8; ++r) { m_run[r] = -1e30f; l_run[r] = 0.f; }

    stage(0, 0);                                  // prologue DMA

    for (int kb = 0; kb < S / 64; ++kb) {
        const int cur = kb & 1;
        wait_async();                             // this wave's DMA done
        __syncthreads();                          // every wave's DMA done
        if (kb + 1 < S / 64) stage(kb + 1, cur ^ 1);  // overlap next DMA

        // ---- scores S = Q K^T (contraction over d, chunks of 32) ----
        v8f sc[4] = {};
#pragma unroll
        for (int j = 0; j < 4; ++j) {
            const int key = j * 16 + l16;
#pragma unroll
            for (int c = 0; c < 2; ++c) {
                HFrag bf;
                bf.h[0] = *(const v8h*)&Ks[cur][key][c * 32 + hi * 8];
                bf.h[1] = *(const v8h*)&Ks[cur][key][c * 32 + 16 + hi * 8];
                sc[j] = WMMA_F16(qa[c].v, bf.v, sc[j]);
            }
        }

        // ---- online softmax (rows live in 16-lane halves of the wave) ----
        float mnew[8], scale[8], rsum[8];
#pragma unroll
        for (int r = 0; r < 8; ++r) {
            float mx = sc[0][r];
#pragma unroll
            for (int j = 1; j < 4; ++j) mx = fmaxf(mx, sc[j][r]);
            mx = fmaxf(mx, __shfl_xor(mx, 1, 32));
            mx = fmaxf(mx, __shfl_xor(mx, 2, 32));
            mx = fmaxf(mx, __shfl_xor(mx, 4, 32));
            mx = fmaxf(mx, __shfl_xor(mx, 8, 32));
            mnew[r]  = fmaxf(m_run[r], mx);
            scale[r] = __expf(m_run[r] - mnew[r]);
            rsum[r]  = 0.f;
        }
#pragma unroll
        for (int j = 0; j < 4; ++j)
#pragma unroll
            for (int r = 0; r < 8; ++r) {
                const float p = __expf(sc[j][r] - mnew[r]);
                sc[j][r] = p;
                rsum[r] += p;
            }
#pragma unroll
        for (int r = 0; r < 8; ++r) {
            rsum[r] += __shfl_xor(rsum[r], 1, 32);
            rsum[r] += __shfl_xor(rsum[r], 2, 32);
            rsum[r] += __shfl_xor(rsum[r], 4, 32);
            rsum[r] += __shfl_xor(rsum[r], 8, 32);
            l_run[r] = l_run[r] * scale[r] + rsum[r];
            m_run[r] = mnew[r];
        }
#pragma unroll
        for (int n = 0; n < 4; ++n)
#pragma unroll
            for (int r = 0; r < 8; ++r) oacc[n][r] *= scale[r];

        // ---- P: C-layout -> LDS -> A-layout fragments ----
#pragma unroll
        for (int j = 0; j < 4; ++j)
#pragma unroll
            for (int r = 0; r < 8; ++r)
                Ps[wave][r + hi * 8][j * 16 + l16] = (_Float16)sc[j][r];

        HFrag pa[2];
#pragma unroll
        for (int c = 0; c < 2; ++c) {
            pa[c].h[0] = *(const v8h*)&Ps[wave][l16][c * 32 + hi * 8];
            pa[c].h[1] = *(const v8h*)&Ps[wave][l16][c * 32 + 16 + hi * 8];
        }

        // ---- O += P @ V  (V^T tile in LDS, shared by all waves) ----
#pragma unroll
        for (int n = 0; n < 4; ++n) {
            const int d = n * 16 + l16;
#pragma unroll
            for (int c = 0; c < 2; ++c) {
                HFrag vb;
                vb.h[0] = *(const v8h*)&Vs[cur][d][c * 32 + hi * 8];
                vb.h[1] = *(const v8h*)&Vs[cur][d][c * 32 + 16 + hi * 8];
                oacc[n] = WMMA_F16(pa[c].v, vb.v, oacc[n]);
            }
        }
    }

    // ---- normalize & write concat-heads layout [b, s, h*64+d] as f16 ----
    float inv[8];
#pragma unroll
    for (int r = 0; r < 8; ++r) inv[r] = 1.0f / l_run[r];
#pragma unroll
    for (int n = 0; n < 4; ++n) {
        const int d = n * 16 + l16;
#pragma unroll
        for (int r = 0; r < 8; ++r) {
            const int s = qbase + r + hi * 8;
            attn[((size_t)bI * S + s) * E + hh * Dh + d] =
                (_Float16)(oacc[n][r] * inv[r]);
        }
    }
}

// ---------------------------------------------------------------------------
// Kernel 3: output projection. attn(f16) @ Wo + bo -> fp32 d_out.
// A tile is already f16 -> staged via async global->LDS DMA.
// ---------------------------------------------------------------------------
__global__ __launch_bounds__(256)
void oproj_kernel(const _Float16* __restrict__ aIn, const float* __restrict__ Wo,
                  const float* __restrict__ bo, float* __restrict__ out)
{
    const int m0 = blockIdx.x * 128;
    const int n0 = blockIdx.y * 64;

    __shared__ __align__(16) _Float16 Xs[128][40];
    __shared__ __align__(16) _Float16 Ws[64][40];

    const int tid  = threadIdx.x;
    const int lane = tid & 31;
    const int wave = tid >> 5;
    const int wr   = wave >> 1;
    const int wc   = wave & 1;
    const int l16  = lane & 15;
    const int hi   = lane >> 4;

    v8f acc[2][2] = {};

    for (int kb = 0; kb < E; kb += 32) {
        __syncthreads();
        // --- async DMA the f16 A tile straight into LDS ---
#pragma unroll
        for (int it = 0; it < 2; ++it) {
            const int u   = tid + it * 256;          // 0..511 16B chunks
            const int row = u >> 2;                  // 0..127
            const int c   = (u & 3) * 8;             // 0..24
            async_copy_b128(aIn + (size_t)(m0 + row) * E + kb + c, &Xs[row][c]);
        }
        // --- W tile f32 -> f16, transposed (sync: converts in flight) ---
#pragma unroll
        for (int it = 0; it < 2; ++it) {
            const int u    = tid + it * 256;
            const int krow = u >> 4;
            const int nc   = (u & 15) * 4;
            float4 f = *(const float4*)(Wo + (size_t)(kb + krow) * E + n0 + nc);
            Ws[nc + 0][krow] = (_Float16)f.x;
            Ws[nc + 1][krow] = (_Float16)f.y;
            Ws[nc + 2][krow] = (_Float16)f.z;
            Ws[nc + 3][krow] = (_Float16)f.w;
        }
        wait_async();
        __syncthreads();

        HFrag a[2], b[2];
#pragma unroll
        for (int i = 0; i < 2; ++i) {
            const int row = wr * 32 + i * 16 + l16;
            a[i].h[0] = *(const v8h*)&Xs[row][hi * 8];
            a[i].h[1] = *(const v8h*)&Xs[row][16 + hi * 8];
        }
#pragma unroll
        for (int j = 0; j < 2; ++j) {
            const int col = wc * 32 + j * 16 + l16;
            b[j].h[0] = *(const v8h*)&Ws[col][hi * 8];
            b[j].h[1] = *(const v8h*)&Ws[col][16 + hi * 8];
        }
#pragma unroll
        for (int i = 0; i < 2; ++i)
#pragma unroll
            for (int j = 0; j < 2; ++j)
                acc[i][j] = WMMA_F16(a[i].v, b[j].v, acc[i][j]);
    }

#pragma unroll
    for (int i = 0; i < 2; ++i)
#pragma unroll
        for (int j = 0; j < 2; ++j) {
            const int col = n0 + wc * 32 + j * 16 + l16;
            const float bv_ = bo[col];
#pragma unroll
            for (int r = 0; r < 8; ++r) {
                const int m = m0 + wr * 32 + i * 16 + r + hi * 8;
                out[(size_t)m * E + col] = acc[i][j][r] + bv_;
            }
        }
}

// ---------------------------------------------------------------------------
extern "C" void kernel_launch(void* const* d_in, const int* in_sizes, int n_in,
                              void* d_out, int out_size, void* d_ws, size_t ws_size,
                              hipStream_t stream) {
    (void)in_sizes; (void)n_in; (void)out_size; (void)ws_size;
    const float* x  = (const float*)d_in[0];
    const float* Wq = (const float*)d_in[1];
    const float* bq = (const float*)d_in[2];
    const float* Wk = (const float*)d_in[3];
    const float* bk = (const float*)d_in[4];
    const float* Wv = (const float*)d_in[5];
    const float* bv = (const float*)d_in[6];
    const float* Wo = (const float*)d_in[7];
    const float* bo = (const float*)d_in[8];

    const size_t MAT = (size_t)M * E;          // 8192*512 f16 elements per buffer
    _Float16* qb   = (_Float16*)d_ws;
    _Float16* kb   = qb + MAT;
    _Float16* vtb  = kb + MAT;
    _Float16* atnb = vtb + MAT;

    dim3 gA(M / 128, E / 64, 3);
    qkv_kernel<<<gA, 256, 0, stream>>>(x, Wq, bq, Wk, bk, Wv, bv, qb, kb, vtb);

    dim3 gF(BSZ * H, S / 128);
    flash_kernel<<<gF, 256, 0, stream>>>(qb, kb, vtb, atnb);

    dim3 gO(M / 128, E / 64);
    oproj_kernel<<<gO, 256, 0, stream>>>(atnb, Wo, bo, (float*)d_out);
}
